// HAN_65266323030533
// MI455X (gfx1250) — matrix operand order repaired and checked by reading, not compiled, not run on hardware
//
#include <hip/hip_runtime.h>
#include <stdint.h>

// ---------------------------------------------------------------------------
// HANConv x2 for MI455X (gfx1250, wave32).
//
// Input order (setup_inputs insertion order; nested params flattened in JAX
// pytree order = sorted dict keys at every level):
//   0: x_author [50000,128] f32      1: x_paper [100000,128] f32
//   2: ei_writes [2,500000] i32      3: ei_written [2,500000] i32
//   4: ei_cites  [2,500000] i32
//   5..17  layer1 params, 18..30 layer2 params, each layer:
//     +0..+2 att_dst (writes, cites, written)   +3..+5 att_src (same order)
//     +6 k_lin/b   +7 k_lin/w [256,256]
//     +8 proj/author/b  +9 proj/author/w [K,256]
//     +10 proj/paper/b  +11 proj/paper/w [K,256]
//     +12 q [256]
// Output: author [50000,256] then paper [100000,256], f32, concatenated.
//
// GEMMs run as bf16 WMMA (v_wmma_f32_16x16x32_bf16) with f32 accumulate;
// A tiles are staged into LDS by the Tensor Data Mover (tensor_load_to_lds +
// s_wait_tensorcnt) when the builtin is available.
// ---------------------------------------------------------------------------

#define NA 50000
#define NP 100000
#define NE 500000
#define HID 256
#define NH 8

typedef __attribute__((ext_vector_type(16))) __bf16 v16bf;
typedef __attribute__((ext_vector_type(8)))  float  v8f;
typedef __attribute__((ext_vector_type(4)))  unsigned int v4u;
typedef __attribute__((ext_vector_type(8)))  int v8i;
typedef __attribute__((ext_vector_type(4)))  int v4i;

#if __has_builtin(__builtin_amdgcn_tensor_load_to_lds) && \
    __has_builtin(__builtin_amdgcn_s_wait_tensorcnt)
#define USE_TDM 1
#else
#define USE_TDM 0
#endif

union Frag16 { uint4 u4[2]; v16bf v; };

__device__ __forceinline__ unsigned short f32_to_bf16(float f) {
  unsigned int u = __float_as_uint(f);
  u = u + 0x7FFFu + ((u >> 16) & 1u);          // round-to-nearest-even
  return (unsigned short)(u >> 16);
}

// ------------------------- generic small kernels ---------------------------
__global__ void fill_u32_kernel(unsigned int* p, unsigned int v, long long n) {
  long long i = (long long)blockIdx.x * blockDim.x + threadIdx.x;
  long long s = (long long)gridDim.x * blockDim.x;
  for (; i < n; i += s) p[i] = v;
}

__global__ void cvt_bf16_kernel(const float* __restrict__ in,
                                unsigned short* __restrict__ out, long long n) {
  long long i = (long long)blockIdx.x * blockDim.x + threadIdx.x;
  long long s = (long long)gridDim.x * blockDim.x;
  for (; i < n; i += s) out[i] = f32_to_bf16(in[i]);
}

__global__ void relu_kernel(float* p, long long n) {
  long long i = (long long)blockIdx.x * blockDim.x + threadIdx.x;
  long long s = (long long)gridDim.x * blockDim.x;
  for (; i < n; i += s) { float v = p[i]; p[i] = v > 0.f ? v : 0.f; }
}

// out = act( attn[0]*in0 + attn[1]*in1 )
__global__ void combine_kernel(const float* __restrict__ in0,
                               const float* __restrict__ in1,
                               const float* __restrict__ attn, int R,
                               float* __restrict__ out, long long n, int act) {
  long long i = (long long)blockIdx.x * blockDim.x + threadIdx.x;
  long long s = (long long)gridDim.x * blockDim.x;
  float a0 = attn[0];
  float a1 = (R > 1) ? attn[1] : 0.f;
  for (; i < n; i += s) {
    float v = a0 * in0[i];
    if (R > 1) v += a1 * in1[i];
    if (act) v = v > 0.f ? v : 0.f;
    out[i] = v;
  }
}

__global__ void reduce_sum_kernel(const float* __restrict__ in, long long n,
                                  float* __restrict__ out) {
  __shared__ float sm[256];
  long long i = (long long)blockIdx.x * blockDim.x + threadIdx.x;
  long long s = (long long)gridDim.x * blockDim.x;
  float acc = 0.f;
  for (; i < n; i += s) acc += in[i];
  sm[threadIdx.x] = acc;
  __syncthreads();
  for (int o = 128; o > 0; o >>= 1) {
    if ((int)threadIdx.x < o) sm[threadIdx.x] += sm[threadIdx.x + o];
    __syncthreads();
  }
  if (threadIdx.x == 0) atomicAdd(out, sm[0]);
}

// softmax over R mean scores: attn[r] = softmax(ssum[r]*invN)
__global__ void semantic_attn_kernel(const float* __restrict__ ssum,
                                     float* __restrict__ attn, int R,
                                     float invN) {
  if (threadIdx.x != 0 || blockIdx.x != 0) return;
  float s[2], m = -3.4e38f;
  for (int r = 0; r < R; ++r) { s[r] = ssum[r] * invN; m = s[r] > m ? s[r] : m; }
  float den = 0.f;
  for (int r = 0; r < R; ++r) { s[r] = __expf(s[r] - m); den += s[r]; }
  for (int r = 0; r < R; ++r) attn[r] = s[r] / den;
}

// ----------------------------- WMMA GEMM -----------------------------------
// C[N,256] = act( Ab[N,K](bf16) @ Wb[K,256](bf16) + bias )
// act==0: plain.  act==2: C = q[col] * tanh(val)   (semantic-score path)
// Block: 256 threads (8 waves), tile 32 rows x 256 cols, K-step 32.
// A tile staged into LDS by the TDM (wave 0 issues tensor_load_to_lds with a
// D# that pads each 64B row by 16B -> LDA=40 u16 stride, zero-fills OOB rows).
#define LDA 40   // padded k-stride (u16 units), multiple of 8 -> 16B aligned frags
__global__ __launch_bounds__(256) void wmma_gemm_kernel(
    const unsigned short* __restrict__ Ab, const unsigned short* __restrict__ Wb,
    const float* __restrict__ bias, const float* __restrict__ qv,
    float* __restrict__ C, int N, int K, int act) {
  __shared__ __align__(16) unsigned short ldsA[32 * LDA];    // [row][k]
  __shared__ __align__(16) unsigned short ldsB[256 * LDA];   // [col][k] (transposed)

  const int tid  = threadIdx.x;
  const int wid  = tid >> 5;
  const int lane = tid & 31;
  const int cg   = wid & 3;        // column group: cols cg*64 .. cg*64+63
  const int rt   = wid >> 2;       // row tile: rows rt*16 .. rt*16+15
  const int l16  = lane & 15;
  const int half = lane >> 4;
  const int blockRow = blockIdx.x * 32;

  v8f acc[4];
  acc[0] = v8f{}; acc[1] = v8f{}; acc[2] = v8f{}; acc[3] = v8f{};

  for (int k0 = 0; k0 < K; k0 += 32) {
    // ---- stage A tile (32 rows x 32 bf16) ----
#if USE_TDM
    if (wid == 0) {
      // D# group0: count=1 | lds_addr | global_addr(57b) | type=2
      unsigned long long ga =
          (unsigned long long)(const void*)(Ab + (unsigned long long)blockRow * K + k0);
      unsigned int ldsoff = (unsigned int)(unsigned long long)(void*)&ldsA[0];
      unsigned int rowsRem = (unsigned int)(N - blockRow);   // tensor_dim1 (OOB rows -> 0)
      v4u g0;
      g0[0] = 1u;                                   // count=1, gather off
      g0[1] = ldsoff;                               // lds_addr
      g0[2] = (unsigned int)ga;                     // global_addr[31:0]
      g0[3] = (unsigned int)(ga >> 32) | (2u << 30);// global_addr[56:32] | type=2
      // D# group1
      v8i g1;
      g1[0] = (int)((1u << 16)      // data_size = 2 bytes
                  | (1u << 20)      // pad_enable
                  | (3u << 22)      // pad_interval: 16 DWORDs (= 64B tile row)
                  | (3u << 25));    // pad_amount: 4 DWORDs (= 16B -> LDA 40)
      g1[1] = (int)(32u << 16);                         // tensor_dim0[15:0] = 32
      g1[2] = (int)((rowsRem & 0xFFFFu) << 16);         // dim0 hi=0 | tensor_dim1 lo16
      g1[3] = (int)((rowsRem >> 16) | (32u << 16));     // tensor_dim1 hi16 | tile_dim0=32
      g1[4] = 32;                                       // tile_dim1=32, tile_dim2=0
      g1[5] = K;                                        // tensor_dim0_stride[31:0]
      g1[6] = 0;
      g1[7] = 0;
      v4i z4 = {0, 0, 0, 0};
#if defined(__clang_major__) && (__clang_major__ >= 23)
      v8i z8 = {0, 0, 0, 0, 0, 0, 0, 0};
      __builtin_amdgcn_tensor_load_to_lds(g0, g1, z4, z4, z8, 0);
#else
      __builtin_amdgcn_tensor_load_to_lds(g0, g1, z4, z4, 0);
#endif
      __builtin_amdgcn_s_wait_tensorcnt(0);
    }
#else
    {
      int r  = tid >> 3;            // 0..31
      int c4 = (tid & 7) * 4;       // 0,4,...,28 (u16 elems)
      int row = blockRow + r;
      uint2 v{};
      if (row < N) v = *(const uint2*)&Ab[(long long)row * K + k0 + c4];
      *(uint2*)&ldsA[r * LDA + c4] = v;
    }
#endif
    // ---- stage W tile transposed: ldsB[col][k] = Wb[k0+k][col] ----
    for (int i = tid; i < 32 * 256; i += 256) {
      int kk = i >> 8, c = i & 255;
      ldsB[c * LDA + kk] = Wb[(long long)(k0 + kk) * 256 + c];
    }
    // prefetch next A tile (global_prefetch_b8)
    if (k0 + 32 < K) {
      int row = blockRow + (tid & 31);
      if (row < N)
        __builtin_prefetch(&Ab[(long long)row * K + k0 + 32 + ((tid >> 5) << 2)], 0, 1);
    }
    __syncthreads();

    // A fragment: 16x32 bf16 per ISA layout
    //   lane half 0: VGPR0-3 k=0..7, VGPR4-7 k=16..23
    //   lane half 1: VGPR0-3 k=8..15, VGPR4-7 k=24..31
    Frag16 a;
    const uint4* pa = (const uint4*)&ldsA[(rt * 16 + l16) * LDA + half * 8];
    a.u4[0] = pa[0];        // elems 0..7  -> k = half*8 + e
    a.u4[1] = pa[2];        // elems 8..15 -> k = 16 + half*8 + (e-8)

#pragma unroll
    for (int t = 0; t < 4; ++t) {
      // B fragment: 32x16 bf16, elem e -> k = half*8 + e (e<8), 16 + half*8 + (e-8) (e>=8), col = l16
      int col = cg * 64 + t * 16 + l16;
      Frag16 b;
      const uint4* pb = (const uint4*)&ldsB[col * LDA + half * 8];
      b.u4[0] = pb[0];
      b.u4[1] = pb[2];
      acc[t] = __builtin_amdgcn_wmma_f32_16x16x32_bf16(
          false, a.v, false, b.v, (short)0, acc[t], false, false);
    }
    __syncthreads();
  }

  // Epilogue: C layout f32 16x16: lane half h holds rows r+8h, col = l16
#pragma unroll
  for (int t = 0; t < 4; ++t) {
    int col = cg * 64 + t * 16 + l16;
    float bb = bias[col];
#pragma unroll
    for (int r = 0; r < 8; ++r) {
      int row = blockRow + rt * 16 + half * 8 + r;
      if (row < N) {
        float v = acc[t][r] + bb;
        if (act == 2) v = qv[col] * tanhf(v);
        C[(long long)row * 256 + col] = v;
      }
    }
  }
}

// ------------------------- edge / attention kernels ------------------------
// a[n,h] = sum_d h[n, h*32+d] * att[h*32+d]
__global__ void attdot_kernel(const float* __restrict__ h,
                              const float* __restrict__ att,
                              float* __restrict__ out, int N) {
  long long idx = (long long)blockIdx.x * blockDim.x + threadIdx.x;
  if (idx >= (long long)N * NH) return;
  int n = (int)(idx >> 3), hh = (int)(idx & 7);
  const float* hp = h + (long long)n * HID + hh * 32;
  const float* ap = att + hh * 32;
  float s = 0.f;
#pragma unroll
  for (int d = 0; d < 32; ++d) s += hp[d] * ap[d];
  out[idx] = s;
}

__device__ __forceinline__ void atomicMaxF32(float* addr, float v) {
  if (v >= 0.f) atomicMax((int*)addr, __float_as_int(v));
  else          atomicMin((unsigned int*)addr, __float_as_uint(v));
}

// alpha = leaky_relu(a_src[row]+a_dst[col]); segment max via atomics
__global__ void alpha_kernel(const int* __restrict__ row,
                             const int* __restrict__ col,
                             const float* __restrict__ aS,
                             const float* __restrict__ aD,
                             float* __restrict__ alphaE,
                             float* __restrict__ amax, int E) {
  long long idx = (long long)blockIdx.x * blockDim.x + threadIdx.x;
  if (idx >= (long long)E * NH) return;
  int e = (int)(idx >> 3), hh = (int)(idx & 7);
  int r = row[e], c = col[e];
  float a = aS[r * NH + hh] + aD[c * NH + hh];
  a = (a >= 0.f) ? a : 0.2f * a;
  alphaE[idx] = a;
  atomicMaxF32(&amax[c * NH + hh], a);
}

// ex = exp(alpha - amax[col]); segment sum via atomics (in-place on alphaE)
__global__ void expsum_kernel(const int* __restrict__ col,
                              const float* __restrict__ amax,
                              float* __restrict__ alphaE,
                              float* __restrict__ den, int E) {
  long long idx = (long long)blockIdx.x * blockDim.x + threadIdx.x;
  if (idx >= (long long)E * NH) return;
  int e = (int)(idx >> 3), hh = (int)(idx & 7);
  int c = col[e];
  float ex = __expf(alphaE[idx] - amax[c * NH + hh]);
  alphaE[idx] = ex;
  atomicAdd(&den[c * NH + hh], ex);
}

// one wave per edge: out[col] += h_src[row] * w   (w per-head, 8 heads x 32 dims)
__global__ __launch_bounds__(256) void msg_kernel(
    const int* __restrict__ row, const int* __restrict__ col,
    const float* __restrict__ alphaE, const float* __restrict__ den,
    const float* __restrict__ hsrc, float* __restrict__ out, int E) {
  int wid = threadIdx.x >> 5, lane = threadIdx.x & 31;
  int e = blockIdx.x * 8 + wid;
  if (e >= E) return;
  int r = row[e], c = col[e];
  float w[NH];
#pragma unroll
  for (int i = 0; i < NH; ++i)
    w[i] = alphaE[(long long)e * NH + i] / den[c * NH + i];
  const float* hs = hsrc + (long long)r * HID;
  float* o = out + (long long)c * HID;
#pragma unroll
  for (int i = 0; i < NH; ++i) {
    int f = i * 32 + lane;
    atomicAdd(&o[f], hs[f] * w[i]);
  }
}

// ------------------------------- launcher ----------------------------------
static inline int gsz(long long n, int b) {
  long long g = (n + b - 1) / b;
  return (int)(g > 1048576 ? 1048576 : g);
}

extern "C" void kernel_launch(void* const* d_in, const int* in_sizes, int n_in,
                              void* d_out, int out_size, void* d_ws, size_t ws_size,
                              hipStream_t stream) {
  (void)in_sizes; (void)n_in; (void)out_size; (void)ws_size;

  const float* x_author = (const float*)d_in[0];
  const float* x_paper  = (const float*)d_in[1];
  const int* ei_writes  = (const int*)d_in[2];
  const int* ei_written = (const int*)d_in[3];
  const int* ei_cites   = (const int*)d_in[4];

  // workspace carving
  size_t off = 0;
  auto allocb = [&](size_t bytes) -> void* {
    void* p = (char*)d_ws + off;
    off = (off + bytes + 255) & ~(size_t)255;
    return p;
  };
  float* hA    = (float*)allocb((size_t)NA * HID * 4);
  float* hP    = (float*)allocb((size_t)NP * HID * 4);
  float* resA  = (float*)allocb((size_t)NA * HID * 4);
  float* resP  = (float*)allocb((size_t)NP * HID * 4);
  float* relA0 = (float*)allocb((size_t)NA * HID * 4);
  float* relP0 = (float*)allocb((size_t)NP * HID * 4);
  float* relP1 = (float*)allocb((size_t)NP * HID * 4);
  float* tbuf  = (float*)allocb((size_t)NP * HID * 4);
  float* aS    = (float*)allocb((size_t)NP * NH * 4);
  float* aD    = (float*)allocb((size_t)NP * NH * 4);
  float* amax  = (float*)allocb((size_t)NP * NH * 4);
  float* den   = (float*)allocb((size_t)NP * NH * 4);
  float* alphaE= (float*)allocb((size_t)NE * NH * 4);
  float* scores= (float*)allocb(16 * 4);               // [0]:A, [4..5]:P, [8..]:attn
  unsigned short* wb0  = (unsigned short*)allocb(256 * 256 * 2);
  unsigned short* wb1  = (unsigned short*)allocb(256 * 256 * 2);
  unsigned short* xAb  = (unsigned short*)allocb((size_t)NA * HID * 2);  // bf16 A (author)
  unsigned short* xPb  = (unsigned short*)allocb((size_t)NP * HID * 2);  // bf16 A (paper)
  unsigned short* relbf= (unsigned short*)allocb((size_t)NP * HID * 2);  // bf16 A (score gemms)

  auto fillz = [&](void* p, long long n32) {
    fill_u32_kernel<<<gsz(n32, 256), 256, 0, stream>>>((unsigned int*)p, 0u, n32);
  };
  auto cvt = [&](const float* src, unsigned short* dst, long long n) {
    cvt_bf16_kernel<<<gsz(n, 256), 256, 0, stream>>>(src, dst, n);
  };
  auto gemm = [&](const unsigned short* Ab, const unsigned short* Wb,
                  const float* bias, const float* qv, float* C, int N, int K,
                  int act) {
    wmma_gemm_kernel<<<(N + 31) / 32, 256, 0, stream>>>(Ab, Wb, bias, qv, C, N, K, act);
  };

  // one HANConv layer
  auto run_layer = [&](int pb, const float* xA, const float* xP, int K,
                       float* outA, float* outP, int actCombine) {
    const float* attD_writes  = (const float*)d_in[pb + 0];
    const float* attD_cites   = (const float*)d_in[pb + 1];
    const float* attD_written = (const float*)d_in[pb + 2];
    const float* attS_writes  = (const float*)d_in[pb + 3];
    const float* attS_cites   = (const float*)d_in[pb + 4];
    const float* attS_written = (const float*)d_in[pb + 5];
    const float* klin_b  = (const float*)d_in[pb + 6];
    const float* klin_w  = (const float*)d_in[pb + 7];
    const float* projA_b = (const float*)d_in[pb + 8];
    const float* projA_w = (const float*)d_in[pb + 9];
    const float* projP_b = (const float*)d_in[pb + 10];
    const float* projP_w = (const float*)d_in[pb + 11];
    const float* qv      = (const float*)d_in[pb + 12];

    // node-type projections (bf16 WMMA, TDM-staged A tiles)
    cvt(xA, xAb, (long long)NA * K);
    cvt(xP, xPb, (long long)NP * K);
    cvt(projA_w, wb0, (long long)K * HID);
    gemm(xAb, wb0, projA_b, nullptr, hA, NA, K, 0);
    cvt(projP_w, wb0, (long long)K * HID);
    gemm(xPb, wb0, projP_b, nullptr, hP, NP, K, 0);

    auto run_rel = [&](const int* ei, const float* hsrc, int Ns,
                       const float* hdst, int Nd, const float* attS,
                       const float* attD, float* relOut) {
      const int* rowp = ei;
      const int* colp = ei + NE;
      attdot_kernel<<<gsz((long long)Ns * NH, 256), 256, 0, stream>>>(hsrc, attS, aS, Ns);
      attdot_kernel<<<gsz((long long)Nd * NH, 256), 256, 0, stream>>>(hdst, attD, aD, Nd);
      fill_u32_kernel<<<gsz((long long)Nd * NH, 256), 256, 0, stream>>>(
          (unsigned int*)amax, 0xFF800000u, (long long)Nd * NH);   // -inf
      fillz(den, (long long)Nd * NH);
      fillz(relOut, (long long)Nd * HID);
      alpha_kernel<<<gsz((long long)NE * NH, 256), 256, 0, stream>>>(
          rowp, colp, aS, aD, alphaE, amax, NE);
      expsum_kernel<<<gsz((long long)NE * NH, 256), 256, 0, stream>>>(
          colp, amax, alphaE, den, NE);
      msg_kernel<<<(NE + 7) / 8, 256, 0, stream>>>(rowp, colp, alphaE, den,
                                                   hsrc, relOut, NE);
      relu_kernel<<<gsz((long long)Nd * HID, 256), 256, 0, stream>>>(
          relOut, (long long)Nd * HID);
    };
    run_rel(ei_writes,  hA, NA, hP, NP, attS_writes,  attD_writes,  relP0);
    run_rel(ei_written, hP, NP, hA, NA, attS_written, attD_written, relA0);
    run_rel(ei_cites,   hP, NP, hP, NP, attS_cites,   attD_cites,   relP1);

    // semantic attention: score_r = mean_n sum_f q_f * tanh((rel_r @ kW + kb))
    cvt(klin_w, wb1, 256 * 256);
    fillz(scores, 16);
    cvt(relA0, relbf, (long long)NA * HID);
    gemm(relbf, wb1, klin_b, qv, tbuf, NA, 256, 2);
    reduce_sum_kernel<<<1024, 256, 0, stream>>>(tbuf, (long long)NA * HID, &scores[0]);
    cvt(relP0, relbf, (long long)NP * HID);
    gemm(relbf, wb1, klin_b, qv, tbuf, NP, 256, 2);
    reduce_sum_kernel<<<1024, 256, 0, stream>>>(tbuf, (long long)NP * HID, &scores[4]);
    cvt(relP1, relbf, (long long)NP * HID);
    gemm(relbf, wb1, klin_b, qv, tbuf, NP, 256, 2);
    reduce_sum_kernel<<<1024, 256, 0, stream>>>(tbuf, (long long)NP * HID, &scores[5]);

    semantic_attn_kernel<<<1, 1, 0, stream>>>(&scores[0], &scores[8], 1, 1.f / NA);
    semantic_attn_kernel<<<1, 1, 0, stream>>>(&scores[4], &scores[12], 2, 1.f / NP);
    combine_kernel<<<gsz((long long)NA * HID, 256), 256, 0, stream>>>(
        relA0, nullptr, &scores[8], 1, outA, (long long)NA * HID, actCombine);
    combine_kernel<<<gsz((long long)NP * HID, 256), 256, 0, stream>>>(
        relP0, relP1, &scores[12], 2, outP, (long long)NP * HID, actCombine);
  };

  float* outA2 = (float*)d_out;
  float* outP2 = (float*)d_out + (long long)NA * HID;

  run_layer(5,  x_author, x_paper, 128, resA, resP, /*relu between layers*/ 1);
  run_layer(18, resA,     resP,    256, outA2, outP2, 0);
}
